// LogModel_89885075570734
// MI455X (gfx1250) — matrix operand order repaired and checked by reading, not compiled, not run on hardware
//
#include <hip/hip_runtime.h>
#include <hip/hip_bf16.h>

typedef _Float16 v16h __attribute__((ext_vector_type(16)));
typedef float    v8f  __attribute__((ext_vector_type(8)));
typedef int      v8i  __attribute__((ext_vector_type(8)));

#define T_LEN 512
#define E_DIM 16
#define H_DIM 32

__device__ __forceinline__ float fast_tanhf(float x) {
#if __has_builtin(__builtin_amdgcn_tanhf)
    return __builtin_amdgcn_tanhf(x);
#elif __has_builtin(__builtin_amdgcn_tanh_f32)
    return __builtin_amdgcn_tanh_f32(x);
#else
    float e = __expf(2.0f * x);
    return 1.0f - 2.0f / (e + 1.0f);
#endif
}
__device__ __forceinline__ float fast_sigmoid(float x) {
    return 0.5f + 0.5f * fast_tanhf(0.5f * x);
}
__device__ __forceinline__ int pack2(float a, float b) {
    auto p = __builtin_amdgcn_cvt_pkrtz(a, b);   // 2 x f16 packed, RTZ
    return __builtin_bit_cast(int, p);
}

// One wave (32 lanes) owns 16 batch rows for the whole T=512 recurrence.
// D[gate,batch] = Wcat[128,49] x [e; h; 1][49,16]  via v_wmma_f32_16x16x32_f16.
//   A (SRC0) = weight tiles, loop-invariant in VGPRs; K=48 column carries bias.
//   B (SRC1) = lane half L: e[8L..8L+7] of batch n, then h[n][8L..8L+7];
//              second WMMA: h[n][16+8L..16+8L+7], then [1,0,...] (bias) in lanes 0-15 only.
// Next-step h operand is lane-local: B takes the same lane's packed h, no swap.
// Loop is blocked by 4 steps: tokens arrive as one int4 per block, embedding
// gathers run on a distance-2 ping-pong (issued >= 1 full step before use).
__global__ __launch_bounds__(32) void lstm_wmma_kernel(
    const int*   __restrict__ x,     const float* __restrict__ emb,
    const float* __restrict__ w_ih,  const float* __restrict__ w_hh,
    const float* __restrict__ b_ih,  const float* __restrict__ b_hh,
    const float* __restrict__ w_fc,  const float* __restrict__ b_fc,
    float* __restrict__ out)
{
    const int lane = threadIdx.x;   // 0..31
    const int n    = lane & 15;     // batch row within tile == B-operand column
    const int L    = lane >> 4;     // K-half of the wave
    const int b0   = blockIdx.x * 16;

    // ---- preload constant A operands: 8 gate tiles x 2 K-slices ----
    v16h A1[8], A2[8];
#pragma unroll
    for (int gt = 0; gt < 8; ++gt) {
        const int gate = gt * 16 + n;                 // A row = l%16
        const float* wi = w_ih + gate * E_DIM;
        const float* wh = w_hh + gate * H_DIM;
        v16h a1, a2;
#pragma unroll
        for (int j = 0; j < 8; ++j) {
            a1[j]     = (_Float16)wi[8 * L + j];          // Kg = 8L+j       (w_ih)
            a1[8 + j] = (_Float16)wh[8 * L + j];          // Kg = 16+8L+j    (w_hh lo)
            a2[j]     = (_Float16)wh[16 + 8 * L + j];     // Kg = 32+8L+j    (w_hh hi)
            a2[8 + j] = (_Float16)0.0f;                   // Kg = 48+8L+j
        }
        if (L == 0) a2[8] = (_Float16)(b_ih[gate] + b_hh[gate]);  // bias @ Kg=48
        A1[gt] = a1; A2[gt] = a2;
    }

    // ---- recurrent state (registers only) ----
    float cS[2][8];                       // cell state: u = 16*up + 8L + r
    float hLoF[8], hHiF[8];               // h as f32 (kept for final FC)
#pragma unroll
    for (int r = 0; r < 8; ++r) { cS[0][r] = 0.f; cS[1][r] = 0.f; hLoF[r] = 0.f; hHiF[r] = 0.f; }
    int hlo[4] = {0, 0, 0, 0};            // packed f16 h, u = 8L + (0..7)
    int hhi[4] = {0, 0, 0, 0};            // packed f16 h, u = 16 + 8L + (0..7)
    const bool isHi = (L != 0);

    // ---- one LSTM step: consume e from (E0..E3), then overwrite them with a
    //      distance-2 prefetch for tokPre, then WMMA + gate math ----
    auto step = [&](float4& E0, float4& E1, float4& E2, float4& E3, int tokPre) {
        // 1) convert current embedding (prefetched >= 1 full step ago)
        int epk[8];
        epk[0] = pack2(E0.x, E0.y); epk[1] = pack2(E0.z, E0.w);
        epk[2] = pack2(E1.x, E1.y); epk[3] = pack2(E1.z, E1.w);
        epk[4] = pack2(E2.x, E2.y); epk[5] = pack2(E2.z, E2.w);
        epk[6] = pack2(E3.x, E3.y); epk[7] = pack2(E3.z, E3.w);

        // 2) issue the distance-2 gather into the (now consumed) buffer
        {
            const float4* ep = (const float4*)(emb + (size_t)tokPre * E_DIM);
            E0 = ep[0]; E1 = ep[1]; E2 = ep[2]; E3 = ep[3];
        }

        // 3) build B operands from lane-local h (SWAPX16 results plo/phi are unused)
        int plo[4], phi[4];
#pragma unroll
        for (int j = 0; j < 4; ++j) {
            plo[j] = __builtin_amdgcn_ds_swizzle(hlo[j], 0x401F);
            phi[j] = __builtin_amdgcn_ds_swizzle(hhi[j], 0x401F);
        }
        v8i b1i, b2i;
#pragma unroll
        for (int j = 0; j < 4; ++j) {
            b1i[j]     = isHi ? epk[4 + j] : epk[j];               // e[8..15] | e[0..7]
            b1i[4 + j] = hlo[j];                                   // h[n][8L..8L+7]
            b2i[j]     = hhi[j];                                   // h[n][16+8L..16+8L+7]
            b2i[4 + j] = (!isHi && j == 0) ? 0x00003C00 : 0;       // ones column (bias) at K=48
        }
        v16h B1 = __builtin_bit_cast(v16h, b1i);
        v16h B2 = __builtin_bit_cast(v16h, b2i);

        // 4) 16 x v_wmma_f32_16x16x32_f16 : D = A1*B1 + (A2*B2 + 0)
        v8f acc[8];
#pragma unroll
        for (int gt = 0; gt < 8; ++gt) {
            v8f z = {0.f, 0.f, 0.f, 0.f, 0.f, 0.f, 0.f, 0.f};
            v8f p = __builtin_amdgcn_wmma_f32_16x16x32_f16(
                        false, A2[gt], false, B2, (short)0, z, false, false);
            acc[gt] = __builtin_amdgcn_wmma_f32_16x16x32_f16(
                        false, A1[gt], false, B1, (short)0, p, false, false);
        }

        // 5) lane-local gate math: i(gt0,1) f(gt2,3) g(gt4,5) o(gt6,7)
#pragma unroll
        for (int up = 0; up < 2; ++up) {
#pragma unroll
            for (int r = 0; r < 8; ++r) {
                float iv = fast_sigmoid(acc[0 + up][r]);
                float fv = fast_sigmoid(acc[2 + up][r]);
                float gv = fast_tanhf (acc[4 + up][r]);
                float ov = fast_sigmoid(acc[6 + up][r]);
                float cv = fv * cS[up][r] + iv * gv;
                cS[up][r] = cv;
                float hv = ov * fast_tanhf(cv);
                if (up) hHiF[r] = hv; else hLoF[r] = hv;
            }
        }
#pragma unroll
        for (int j = 0; j < 4; ++j) {
            hlo[j] = pack2(hLoF[2 * j], hLoF[2 * j + 1]);
            hhi[j] = pack2(hHiF[2 * j], hHiF[2 * j + 1]);
        }
    };

    // ---- blocked recurrence: 128 blocks x 4 steps ----
    const int*  xrow  = x + (size_t)(b0 + n) * T_LEN;
    const int4* xrow4 = (const int4*)xrow;

    int4 tokC = xrow4[0];                 // tokens t = 0..3
    float4 Ea0, Ea1, Ea2, Ea3;            // e buffer A (even steps)
    float4 Eb0, Eb1, Eb2, Eb3;            // e buffer B (odd steps)
    {
        const float4* ea = (const float4*)(emb + (size_t)tokC.x * E_DIM);
        Ea0 = ea[0]; Ea1 = ea[1]; Ea2 = ea[2]; Ea3 = ea[3];
        const float4* eb = (const float4*)(emb + (size_t)tokC.y * E_DIM);
        Eb0 = eb[0]; Eb1 = eb[1]; Eb2 = eb[2]; Eb3 = eb[3];
    }

#pragma clang loop unroll(disable)
    for (int tb = 0; tb < T_LEN / 4; ++tb) {
        const int tbn = (tb + 1 > T_LEN / 4 - 1) ? (T_LEN / 4 - 1) : (tb + 1);
        int4 tokN = xrow4[tbn];           // tokens for the next block (in flight)

        step(Ea0, Ea1, Ea2, Ea3, tokC.z); // t = 4tb+0, prefetch t+2
        step(Eb0, Eb1, Eb2, Eb3, tokC.w); // t = 4tb+1, prefetch t+3
        step(Ea0, Ea1, Ea2, Ea3, tokN.x); // t = 4tb+2, prefetch t+4
        step(Eb0, Eb1, Eb2, Eb3, tokN.y); // t = 4tb+3, prefetch t+5

        tokC = tokN;
    }

    // ---- final FC: out[b] = sigmoid(h . w_fc + b_fc) ----
    float s = 0.f;
#pragma unroll
    for (int r = 0; r < 8; ++r) {
        s += hLoF[r] * w_fc[8 * L + r];
        s += hHiF[r] * w_fc[16 + 8 * L + r];
    }
    s += __builtin_bit_cast(float,
            __builtin_amdgcn_ds_swizzle(__builtin_bit_cast(int, s), 0x401F));
    if (lane < 16) out[b0 + n] = fast_sigmoid(s + b_fc[0]);
}

extern "C" void kernel_launch(void* const* d_in, const int* in_sizes, int n_in,
                              void* d_out, int out_size, void* d_ws, size_t ws_size,
                              hipStream_t stream) {
    const int*   x    = (const int*)  d_in[0];
    const float* emb  = (const float*)d_in[1];
    const float* w_ih = (const float*)d_in[2];
    const float* w_hh = (const float*)d_in[3];
    const float* b_ih = (const float*)d_in[4];
    const float* b_hh = (const float*)d_in[5];
    const float* w_fc = (const float*)d_in[6];
    const float* b_fc = (const float*)d_in[7];

    const int B = in_sizes[0] / T_LEN;       // 8192
    lstm_wmma_kernel<<<dim3(B / 16), dim3(32), 0, stream>>>(
        x, emb, w_ih, w_hh, b_ih, b_hh, w_fc, b_fc, (float*)d_out);
}
